// AttentionGuidance_76115410420004
// MI455X (gfx1250) — matrix operand order repaired and checked by reading, not compiled, not run on hardware
//
#include <hip/hip_runtime.h>
#include <hip/hip_bf16.h>

typedef _Float16 f16;
typedef __attribute__((ext_vector_type(16))) _Float16 v16h;
typedef __attribute__((ext_vector_type(8)))  _Float16 v8h;
typedef __attribute__((ext_vector_type(8)))  float    v8f;

#define BATCH 2
#define SEQ   2048
#define DMODEL 1024
#define HEADS 16
#define HDIM  64
#define NTOK  (BATCH*SEQ)          // 4096
#define QK_SCALE 0.125f            // 1/sqrt(64)
#define LDSPAD 72                  // 64 + 8 f16 row padding (144B rows, bank-friendly)

// ---------------- WMMA helpers ----------------

__device__ inline v8f wmma_f16(v16h a, v16h b, v8f c) {
  // D = A(16x32) * B(32x16) + C, f32 accumulate
  return __builtin_amdgcn_wmma_f32_16x16x32_f16(false, a, false, b, (short)0, c, false, false);
}

// A fragment: 16x32 (MxK), A row-major with leading dim lda (f16 elements).
// lanes 0-15 -> M=lane, K = {k0..k0+7, k0+16..k0+23}
// lanes 16-31 -> M=lane-16, K = {k0+8..k0+15, k0+24..k0+31}
__device__ inline v16h load_frag_a(const f16* A, int lda, int m0, int k0) {
  int lane = threadIdx.x & 31;
  int m    = m0 + (lane & 15);
  int hi   = (lane >> 4) & 1;
  const f16* r = A + (size_t)m * lda + k0 + hi * 8;
  v8h lo = *(const v8h*)(r);
  v8h hv = *(const v8h*)(r + 16);
  v16h out;
#pragma unroll
  for (int i = 0; i < 8; ++i) { out[i] = lo[i]; out[8 + i] = hv[i]; }
  return out;
}

// B fragment: 32x16 (KxN), sourced from BT[n][k] (row-major transpose, ld=ldb).
// lanes 0-15 -> N=lane, K = {k0..k0+7, k0+16..k0+23} ; lanes 16-31 -> N=lane-16, K = {k0+8..k0+15, k0+24..k0+31}
__device__ inline v16h load_frag_b(const f16* BT, int ldb, int n0, int k0) {
  int lane = threadIdx.x & 31;
  int n    = n0 + (lane & 15);
  int koff = k0 + ((lane >> 4) & 1) * 8;
  const f16* r = BT + (size_t)n * ldb + koff;
  v8h lo = *(const v8h*)(r);
  v8h hv = *(const v8h*)(r + 16);
  v16h out;
#pragma unroll
  for (int i = 0; i < 8; ++i) { out[i] = lo[i]; out[8 + i] = hv[i]; }
  return out;
}

// ---------------- async memory -> LDS (CDNA5 ASYNCcnt path) ----------------

__device__ inline void async_b128_to_lds(unsigned int lds_byte_off, const f16* g) {
  asm volatile("global_load_async_to_lds_b128 %0, %1, off"
               :: "v"(lds_byte_off), "v"(g) : "memory");
}

__device__ inline void wait_async0() {
  asm volatile("s_wait_asynccnt 0x0" ::: "memory");
}

// Stage a 64-row x 64-f16 tile (row stride grow in f16) into an LDS tile with
// LDSPAD row stride. 512 x 16B transfers, cooperatively by 128 threads.
__device__ inline void stage_tile_async(f16* lds_base, const f16* gbase,
                                        int grow, int tid) {
  unsigned int lbase = (unsigned int)(size_t)lds_base;  // LDS offset = addr[31:0]
#pragma unroll
  for (int it = 0; it < 4; ++it) {
    int t   = it * 128 + tid;
    int row = t >> 3;     // 0..63
    int ch  = t & 7;      // 0..7 (16B chunks)
    unsigned int loff = lbase + (unsigned int)(row * LDSPAD + ch * 8) * 2u;
    async_b128_to_lds(loff, gbase + (size_t)row * grow + ch * 8);
  }
}

// ---------------- 1) fp32 -> f16 convert ----------------

__global__ void cvt_f32_to_f16(const float* __restrict__ src, f16* __restrict__ dst, int n) {
  int i = blockIdx.x * blockDim.x + threadIdx.x;
  int stride = gridDim.x * blockDim.x;
  for (; i < n; i += stride) dst[i] = (f16)src[i];
}

// ---------------- 2) transpose + convert weights: WT[n][k] = W[k][n] ----------------

__global__ void transpose_cvt(const float* __restrict__ W, f16* __restrict__ WT,
                              int rows, int cols) {  // W is rows x cols
  __shared__ float tile[32][33];
  int bx = blockIdx.x * 32;  // col base (n)
  int by = blockIdx.y * 32;  // row base (k)
  int tx = threadIdx.x, ty = threadIdx.y;     // block (32,8)
#pragma unroll
  for (int i = 0; i < 32; i += 8)
    tile[ty + i][tx] = W[(size_t)(by + ty + i) * cols + bx + tx];
  __syncthreads();
#pragma unroll
  for (int i = 0; i < 32; i += 8)
    WT[(size_t)(bx + ty + i) * rows + by + tx] = (f16)tile[tx][ty + i];
}

// ---------------- 3) QKV projections ----------------
// z=0 -> Q[b][h][s][d] (scaled), z=1 -> K[b][h][s][d], z=2 -> Vt[b][h][d][s]
__global__ __launch_bounds__(256) void qkv_gemm(const f16* __restrict__ X,
                                                const f16* __restrict__ WqT,
                                                const f16* __restrict__ WkT,
                                                const f16* __restrict__ WvT,
                                                f16* __restrict__ Q,
                                                f16* __restrict__ K,
                                                f16* __restrict__ Vt) {
  int mode = blockIdx.z;
  const f16* WT  = (mode == 0) ? WqT : (mode == 1) ? WkT : WvT;
  f16*       out = (mode == 0) ? Q   : (mode == 1) ? K   : Vt;

  int wid   = blockIdx.x * 8 + (threadIdx.x >> 5);
  int mTile = wid >> 4;
  int nStrip= wid & 15;
  int m0 = mTile * 16;
  int n0 = nStrip * 64;
  int lane = threadIdx.x & 31;
  int hi   = (lane >> 4) & 1;

  v8f acc[4] = {};
  for (int k0 = 0; k0 < DMODEL; k0 += 32) {
    v16h a = load_frag_a(X, DMODEL, m0, k0);
#pragma unroll
    for (int j = 0; j < 4; ++j) {
      v16h b = load_frag_b(WT, DMODEL, n0 + j * 16, k0);
      acc[j] = wmma_f16(a, b, acc[j]);
    }
  }

  float scale = (mode == 0) ? QK_SCALE : 1.0f;
#pragma unroll
  for (int j = 0; j < 4; ++j) {
    int n = n0 + j * 16 + (lane & 15);
    int h = n >> 6, d = n & 63;
#pragma unroll
    for (int r = 0; r < 8; ++r) {
      int m = m0 + r + 8 * hi;
      int b = m >> 11, s = m & 2047;
      float v = acc[j][r] * scale;
      size_t idx;
      if (mode == 2) idx = (((size_t)(b * HEADS + h) * HDIM + d) * SEQ + s);   // Vt[b][h][d][s]
      else           idx = (((size_t)(b * HEADS + h) * SEQ + s) * HDIM + d);   // Q/K[b][h][s][d]
      out[idx] = (f16)v;
    }
  }
}

// ---------------- 4) Flash attention with async-LDS K/V staging ----------------
// grid: (SEQ/64, BATCH*HEADS), block: 128 (4 waves). Each wave: 16 query rows.
// K/V tiles for each 64-key block are DMA'd once per workgroup into LDS
// (double-buffered) instead of being fetched 4x redundantly per wave.
__global__ __launch_bounds__(128) void flash_attn(const f16* __restrict__ Q,
                                                  const f16* __restrict__ K,
                                                  const f16* __restrict__ Vt,
                                                  f16* __restrict__ O) {
  __shared__ alignas(16) f16 Ktile[2][64][LDSPAD];  // [buf][key][d]
  __shared__ alignas(16) f16 Vtile[2][64][LDSPAD];  // [buf][d][key]
  __shared__ alignas(16) f16 Pbuf[4][16][LDSPAD];   // per-wave P reshape tile

  int bh = blockIdx.y;
  int b  = bh >> 4, h = bh & 15;
  int tid  = threadIdx.x;
  int w    = tid >> 5;
  int lane = tid & 31;
  int hi   = (lane >> 4) & 1;
  int qrow = blockIdx.x * 64 + w * 16;

  const f16* Qp  = Q  + (size_t)bh * SEQ * HDIM;
  const f16* Kp  = K  + (size_t)bh * SEQ * HDIM;
  const f16* Vtp = Vt + (size_t)bh * HDIM * SEQ;

  // Q fragments stay in registers (pre-scaled by 1/sqrt(d))
  v16h aq0 = load_frag_a(Qp, HDIM, qrow, 0);
  v16h aq1 = load_frag_a(Qp, HDIM, qrow, 32);

  float mrow[8], lrow[8];
  v8f acc_o[4] = {};
#pragma unroll
  for (int r = 0; r < 8; ++r) { mrow[r] = -INFINITY; lrow[r] = 0.0f; }

  f16* PW = &Pbuf[w][0][0];

  // prologue: stage key-block 0
  stage_tile_async(&Ktile[0][0][0], Kp,       HDIM, tid);
  stage_tile_async(&Vtile[0][0][0], Vtp,      SEQ,  tid);
  wait_async0();
  __syncthreads();

  int buf = 0;
  for (int kb = 0; kb < SEQ; kb += 64, buf ^= 1) {
    // issue DMA for next key block into the alternate buffer
    if (kb + 64 < SEQ) {
      stage_tile_async(&Ktile[buf ^ 1][0][0], Kp + (size_t)(kb + 64) * HDIM, HDIM, tid);
      stage_tile_async(&Vtile[buf ^ 1][0][0], Vtp + (kb + 64),               SEQ,  tid);
    }
    const f16* Kt = &Ktile[buf][0][0];
    const f16* Vl = &Vtile[buf][0][0];

    // ---- scores: S(16x64) = Q @ K^T (B-operand streamed from LDS) ----
    v8f sacc[4] = {};
#pragma unroll
    for (int j = 0; j < 4; ++j) {
      v16h bk0 = load_frag_b(Kt, LDSPAD, j * 16, 0);
      sacc[j] = wmma_f16(aq0, bk0, sacc[j]);
      v16h bk1 = load_frag_b(Kt, LDSPAD, j * 16, 32);
      sacc[j] = wmma_f16(aq1, bk1, sacc[j]);
    }

    // ---- online softmax (row = r + 8*hi lives across one 16-lane half) ----
    float alpha[8];
#pragma unroll
    for (int r = 0; r < 8; ++r) {
      float rmax = sacc[0][r];
#pragma unroll
      for (int j = 1; j < 4; ++j) rmax = fmaxf(rmax, sacc[j][r]);
#pragma unroll
      for (int off = 1; off < 16; off <<= 1)
        rmax = fmaxf(rmax, __shfl_xor(rmax, off, 32));
      float nm = fmaxf(mrow[r], rmax);
      alpha[r] = __expf(mrow[r] - nm);
      mrow[r]  = nm;
    }
#pragma unroll
    for (int j = 0; j < 4; ++j)
#pragma unroll
      for (int r = 0; r < 8; ++r)
        sacc[j][r] = __expf(sacc[j][r] - mrow[r]);
#pragma unroll
    for (int r = 0; r < 8; ++r) {
      float rsum = sacc[0][r] + sacc[1][r] + sacc[2][r] + sacc[3][r];
#pragma unroll
      for (int off = 1; off < 16; off <<= 1)
        rsum += __shfl_xor(rsum, off, 32);
      lrow[r] = lrow[r] * alpha[r] + rsum;
    }
#pragma unroll
    for (int j = 0; j < 4; ++j)
#pragma unroll
      for (int r = 0; r < 8; ++r)
        acc_o[j][r] *= alpha[r];

    // ---- C-layout -> A-fragment reshape through per-wave LDS tile ----
#pragma unroll
    for (int j = 0; j < 4; ++j) {
      int col = j * 16 + (lane & 15);
#pragma unroll
      for (int r = 0; r < 8; ++r)
        Pbuf[w][r + 8 * hi][col] = (f16)sacc[j][r];
    }
    // DS ops are in-order within a wave; wave-private tile -> no barrier
    v16h ap0 = load_frag_a(PW, LDSPAD, 0, 0);
    v16h ap1 = load_frag_a(PW, LDSPAD, 0, 32);

    // ---- O += P @ V (V tile from LDS) ----
#pragma unroll
    for (int j = 0; j < 4; ++j) {
      v16h bv0 = load_frag_b(Vl, LDSPAD, j * 16, 0);
      acc_o[j] = wmma_f16(ap0, bv0, acc_o[j]);
      v16h bv1 = load_frag_b(Vl, LDSPAD, j * 16, 32);
      acc_o[j] = wmma_f16(ap1, bv1, acc_o[j]);
    }

    // next-tile DMA must have landed, and everyone must be done with `buf`
    wait_async0();
    __syncthreads();
  }

  // ---- normalize + store O[b][s][h*64+d] ----
#pragma unroll
  for (int r = 0; r < 8; ++r) {
    float inv = 1.0f / lrow[r];
    int srow = qrow + r + 8 * hi;
#pragma unroll
    for (int j = 0; j < 4; ++j) {
      int col = h * HDIM + j * 16 + (lane & 15);
      O[((size_t)b * SEQ + srow) * DMODEL + col] = (f16)(acc_o[j][r] * inv);
    }
  }
}

// ---------------- 5) output projection + bias ----------------

__global__ __launch_bounds__(256) void out_proj(const f16* __restrict__ A,
                                                const f16* __restrict__ WoT,
                                                const float* __restrict__ bo,
                                                float* __restrict__ out) {
  int wid   = blockIdx.x * 8 + (threadIdx.x >> 5);
  int mTile = wid >> 4;
  int nStrip= wid & 15;
  int m0 = mTile * 16;
  int n0 = nStrip * 64;
  int lane = threadIdx.x & 31;
  int hi   = (lane >> 4) & 1;

  v8f acc[4] = {};
  for (int k0 = 0; k0 < DMODEL; k0 += 32) {
    v16h a = load_frag_a(A, DMODEL, m0, k0);
#pragma unroll
    for (int j = 0; j < 4; ++j) {
      v16h b = load_frag_b(WoT, DMODEL, n0 + j * 16, k0);
      acc[j] = wmma_f16(a, b, acc[j]);
    }
  }
#pragma unroll
  for (int j = 0; j < 4; ++j) {
    int n = n0 + j * 16 + (lane & 15);
    float bias = bo[n];
#pragma unroll
    for (int r = 0; r < 8; ++r) {
      int m = m0 + r + 8 * hi;
      out[(size_t)m * DMODEL + n] = acc[j][r] + bias;
    }
  }
}

// ---------------- host launch ----------------

extern "C" void kernel_launch(void* const* d_in, const int* in_sizes, int n_in,
                              void* d_out, int out_size, void* d_ws, size_t ws_size,
                              hipStream_t stream) {
  const float* hs = (const float*)d_in[0];
  const float* Wq = (const float*)d_in[1];
  const float* Wk = (const float*)d_in[2];
  const float* Wv = (const float*)d_in[3];
  const float* Wo = (const float*)d_in[4];
  const float* bo = (const float*)d_in[5];

  char* ws = (char*)d_ws;
  const size_t MB = 1ull << 20;
  f16* Xh  = (f16*)(ws + 0 * MB);    // 8 MB : hidden f16
  f16* WqT = (f16*)(ws + 8 * MB);    // 2 MB
  f16* WkT = (f16*)(ws + 10 * MB);   // 2 MB
  f16* WvT = (f16*)(ws + 12 * MB);   // 2 MB
  f16* WoT = (f16*)(ws + 14 * MB);   // 2 MB
  f16* Qb  = (f16*)(ws + 16 * MB);   // 8 MB : Q[b][h][s][d] (pre-scaled)
  f16* Kb  = (f16*)(ws + 24 * MB);   // 8 MB : K[b][h][s][d]
  f16* Vt  = (f16*)(ws + 32 * MB);   // 8 MB : Vt[b][h][d][s]
  f16* Ob  = (f16*)(ws + 40 * MB);   // 8 MB : attention output f16
  // total 48 MB of workspace

  cvt_f32_to_f16<<<2048, 256, 0, stream>>>(hs, Xh, NTOK * DMODEL);

  dim3 tb(32, 8), tg(DMODEL / 32, DMODEL / 32);
  transpose_cvt<<<tg, tb, 0, stream>>>(Wq, WqT, DMODEL, DMODEL);
  transpose_cvt<<<tg, tb, 0, stream>>>(Wk, WkT, DMODEL, DMODEL);
  transpose_cvt<<<tg, tb, 0, stream>>>(Wv, WvT, DMODEL, DMODEL);
  transpose_cvt<<<tg, tb, 0, stream>>>(Wo, WoT, DMODEL, DMODEL);

  qkv_gemm<<<dim3(512, 1, 3), 256, 0, stream>>>(Xh, WqT, WkT, WvT, Qb, Kb, Vt);

  flash_attn<<<dim3(SEQ / 64, BATCH * HEADS), 128, 0, stream>>>(Qb, Kb, Vt, Ob);

  out_proj<<<512, 256, 0, stream>>>(Ob, WoT, bo, (float*)d_out);
}
